// TopKSAE_386547057040
// MI455X (gfx1250) — matrix operand order repaired and checked by reading, not compiled, not run on hardware
//
#include <hip/hip_runtime.h>
#include <hip/hip_bf16.h>

// ---------------------------------------------------------------------------
// TopK SAE for MI455X (gfx1250, wave32, WMMA)
//   encode : z_pre = relu(x @ W_enc^T + b_enc)   [8192 x 16384]  (bf16-split WMMA,
//            double-buffered LDS, global prefetch overlapped with WMMA)
//   topk   : keep top-64 per row (exact radix select), zero the rest
//   decode : x_hat = b_dec + sum_k val_k * W_enc[idx_k, :]  (W_dec == W_enc^T)
// ---------------------------------------------------------------------------

#define BATCH   8192
#define IN_DIM  2048
#define LAT     16384
#define KTOP    64

typedef __attribute__((ext_vector_type(16))) __bf16 v16bf;
typedef __attribute__((ext_vector_type(8)))  __bf16 v8bf;
typedef __attribute__((ext_vector_type(4)))  __bf16 v4bf;
typedef __attribute__((ext_vector_type(8)))  float  v8f;

#define TM 128          // block M tile
#define TN 128          // block N tile
#define TK 32           // K chunk (one bf16 WMMA depth)
#define LDSS 40         // padded LDS row stride in bf16 elements (80B -> conflict-free)

// Load a v16bf A-fragment: lane half 0 -> K {0..7, 16..23}, half 1 -> {8..15, 24..31}
__device__ __forceinline__ v16bf ldfragA(const __bf16* p) {
    v8bf a = *(const v8bf*)(p);        // K base + 0..7
    v8bf b = *(const v8bf*)(p + 16);   // K base + 16..23
    return __builtin_shufflevector(a, b, 0,1,2,3,4,5,6,7,8,9,10,11,12,13,14,15);
}
// Load a v16bf B-fragment: lane half 0 -> K {0..7, 16..23}, half 1 -> {8..15, 24..31}
__device__ __forceinline__ v16bf ldfragB(const __bf16* p) {
    v8bf a = *(const v8bf*)(p);        // K base + 0..7
    v8bf b = *(const v8bf*)(p + 16);   // K base + 16..23
    return __builtin_shufflevector(a, b, 0,1,2,3,4,5,6,7,8,9,10,11,12,13,14,15);
}

// f32 -> (hi, lo) bf16 split store (4 elements) into LDS
__device__ __forceinline__ void cvt_store4(__bf16* hi, __bf16* lo, int e, float4 f) {
    float ff[4] = {f.x, f.y, f.z, f.w};
    v4bf h, l;
#pragma unroll
    for (int t = 0; t < 4; ++t) {
        __bf16 hh = (__bf16)ff[t];
        h[t] = hh;
        l[t] = (__bf16)(ff[t] - (float)hh);
    }
    *(v4bf*)(hi + e) = h;
    *(v4bf*)(lo + e) = l;
}

// ---------------------------------------------------------------------------
// Kernel 1: encoder GEMM + bias + relu, writes pre_acts into z region of d_out.
// Double-buffered LDS; next K-chunk's global loads issued before the WMMA chain
// so VMEM latency hides under matrix math; one barrier per K iteration.
// ---------------------------------------------------------------------------
__global__ __launch_bounds__(256)
void sae_encode(const float* __restrict__ x, const float* __restrict__ W,
                const float* __restrict__ b_enc, float* __restrict__ z)
{
    __shared__ __attribute__((aligned(16))) __bf16 Ah[2][TM * LDSS];
    __shared__ __attribute__((aligned(16))) __bf16 Al[2][TM * LDSS];
    __shared__ __attribute__((aligned(16))) __bf16 Bh[2][TN * LDSS];
    __shared__ __attribute__((aligned(16))) __bf16 Bl[2][TN * LDSS];

    const int tid   = threadIdx.x;
    const int lane  = tid & 31;
    const int wave  = tid >> 5;          // 8 waves
    const int waveM = wave >> 2;         // 0..1  (64-row strip)
    const int waveN = wave & 3;          // 0..3  (32-col strip)
    const int n0 = blockIdx.x * TN;
    const int m0 = blockIdx.y * TM;

    // tile loader mapping: 256 threads cover a 32-row x 32-col f32 patch per step
    const int lr = tid >> 3;             // 0..31
    const int lc = (tid & 7) * 4;        // 0,4,...,28

    v8f acc[4][2] = {};                  // 4 m-tiles x 2 n-tiles of 16x16 f32

    const int r16   = lane & 15;
    const int half  = lane >> 4;
    const int kA    = half * 8;          // A-fragment K sub-offset
    const int kB    = half * 8;          // B-fragment K sub-offset

    float4 pa[4], pb[4];                 // global prefetch registers

    auto load_chunk = [&](int k0) {
#pragma unroll
        for (int it = 0; it < 4; ++it) {
            const int r = lr + it * 32;
            pa[it] = *(const float4*)(x + (size_t)(m0 + r) * IN_DIM + k0 + lc);
            pb[it] = *(const float4*)(W + (size_t)(n0 + r) * IN_DIM + k0 + lc);
        }
    };
    auto store_chunk = [&](int b) {
#pragma unroll
        for (int it = 0; it < 4; ++it) {
            const int r = lr + it * 32;
            cvt_store4(Ah[b], Al[b], r * LDSS + lc, pa[it]);
            cvt_store4(Bh[b], Bl[b], r * LDSS + lc, pb[it]);
        }
    };

    load_chunk(0);
    store_chunk(0);
    __syncthreads();

    int cur = 0;
#pragma unroll 1
    for (int k0 = 0; k0 < IN_DIM; k0 += TK) {
        const bool has_next = (k0 + TK) < IN_DIM;   // uniform -> scalar branch
        if (has_next) load_chunk(k0 + TK);          // issue VMEM early

        v16bf ah[4], al[4], bh[2], bl[2];
#pragma unroll
        for (int i = 0; i < 4; ++i) {
            const int base = (waveM * 64 + i * 16 + r16) * LDSS + kA;
            ah[i] = ldfragA(Ah[cur] + base);
            al[i] = ldfragA(Al[cur] + base);
        }
#pragma unroll
        for (int j = 0; j < 2; ++j) {
            const int base = (waveN * 32 + j * 16 + r16) * LDSS + kB;
            bh[j] = ldfragB(Bh[cur] + base);
            bl[j] = ldfragB(Bl[cur] + base);
        }
#pragma unroll
        for (int i = 0; i < 4; ++i)
#pragma unroll
            for (int j = 0; j < 2; ++j) {
                // split-bf16: hi*hi + hi*lo + lo*hi ~= fp32 product
                acc[i][j] = __builtin_amdgcn_wmma_f32_16x16x32_bf16(
                    false, ah[i], false, bh[j], (short)0, acc[i][j], false, false);
                acc[i][j] = __builtin_amdgcn_wmma_f32_16x16x32_bf16(
                    false, ah[i], false, bl[j], (short)0, acc[i][j], false, false);
                acc[i][j] = __builtin_amdgcn_wmma_f32_16x16x32_bf16(
                    false, al[i], false, bh[j], (short)0, acc[i][j], false, false);
            }

        if (has_next) store_chunk(cur ^ 1);         // fill the other buffer
        __syncthreads();
        cur ^= 1;
    }

    // epilogue: C layout -> VGPR r: lanes0-15 M=r, lanes16-31 M=r+8; N=lane&15
    float bias[2];
#pragma unroll
    for (int j = 0; j < 2; ++j)
        bias[j] = b_enc[n0 + waveN * 32 + j * 16 + (lane & 15)];

#pragma unroll
    for (int i = 0; i < 4; ++i)
#pragma unroll
        for (int j = 0; j < 2; ++j) {
            const int col = n0 + waveN * 32 + j * 16 + (lane & 15);
#pragma unroll
            for (int r = 0; r < 8; ++r) {
                const int row = m0 + waveM * 64 + i * 16 + r + half * 8;
                float v = acc[i][j][r] + bias[j];
                v = fmaxf(v, 0.0f);
                z[(size_t)row * LAT + col] = v;
            }
        }
}

// ---------------------------------------------------------------------------
// Kernel 2: exact per-row top-64 via 4-pass radix select (values >= 0 after relu,
// so uint compare == float compare). Zeroes non-topk in z; compacts (idx,val)->ws.
// ---------------------------------------------------------------------------
__global__ __launch_bounds__(256)
void sae_topk(float* __restrict__ z, int* __restrict__ idx_ws, float* __restrict__ val_ws)
{
    extern __shared__ unsigned int smem[];
    unsigned* sv   = smem;          // LAT entries (64 KB)
    unsigned* hist = smem + LAT;    // 256 entries
    __shared__ unsigned s_prefix, s_k, s_cntEq, s_slot;

    const int row = blockIdx.x;
    const int tid = threadIdx.x;
    float* zr = z + (size_t)row * LAT;

    for (int i = tid; i < LAT; i += 256) sv[i] = __float_as_uint(zr[i]);
    if (tid == 0) { s_prefix = 0u; s_k = KTOP; s_slot = 0u; }
    if (tid < KTOP) { idx_ws[row * KTOP + tid] = 0; val_ws[row * KTOP + tid] = 0.0f; }
    __syncthreads();

    for (int p = 3; p >= 0; --p) {
        hist[tid] = 0u;
        __syncthreads();
        const unsigned pref = s_prefix;
        for (int i = tid; i < LAT; i += 256) {
            const unsigned u = sv[i];
            const bool match = (p == 3) || ((u >> ((p + 1) * 8)) == pref);
            if (match) atomicAdd(&hist[(u >> (p * 8)) & 255u], 1u);
        }
        __syncthreads();
        if (tid == 0) {
            unsigned k = s_k;
            int b = 255;
            for (; b > 0; --b) { const unsigned c = hist[b]; if (k <= c) break; k -= c; }
            s_prefix = (pref << 8) | (unsigned)b;
            s_k      = k;
            s_cntEq  = hist[b];
        }
        __syncthreads();
    }

    const unsigned T     = s_prefix;   // exact 64th-largest value (bit pattern)
    const unsigned need  = s_k;        // # of == T entries inside top-K
    const unsigned cntEq = s_cntEq;
    const unsigned DROP  = 0xFFFFFFFFu;

    // tie-break: jax top_k keeps lowest indices; serial fixup (measure-zero case)
    if (tid == 0 && T != 0u && cntEq > need) {
        unsigned kept = 0;
        for (int i = 0; i < LAT; ++i)
            if (sv[i] == T) { if (kept < need) ++kept; else sv[i] = DROP; }
    }
    __syncthreads();

    for (int i = tid; i < LAT; i += 256) {
        const unsigned u = sv[i];
        if (u > T) {
            if (u == DROP) { zr[i] = 0.0f; }
            else {
                const unsigned s = atomicAdd(&s_slot, 1u);
                idx_ws[row * KTOP + s] = i;
                val_ws[row * KTOP + s] = __uint_as_float(u);
            }
        } else if (u == T) {
            if (T != 0u) {   // T==0 survivors contribute nothing; leave defaults
                const unsigned s = atomicAdd(&s_slot, 1u);
                idx_ws[row * KTOP + s] = i;
                val_ws[row * KTOP + s] = __uint_as_float(u);
            }
        } else {
            zr[i] = 0.0f;
        }
    }
}

// ---------------------------------------------------------------------------
// Kernel 3: sparse decode. W_dec is constructed as W_enc^T, so column idx of
// W_dec == contiguous row idx of W_enc -> fully coalesced, L2-resident reads.
// ---------------------------------------------------------------------------
__global__ __launch_bounds__(256)
void sae_decode(const float* __restrict__ Wenc, const float* __restrict__ b_dec,
                const int* __restrict__ idx_ws, const float* __restrict__ val_ws,
                float* __restrict__ xhat)
{
    __shared__ int   sidx[KTOP];
    __shared__ float sval[KTOP];
    const int row = blockIdx.x;
    const int tid = threadIdx.x;
    if (tid < KTOP) { sidx[tid] = idx_ws[row * KTOP + tid]; sval[tid] = val_ws[row * KTOP + tid]; }
    __syncthreads();

    const int d0 = tid * 8;     // 256 threads * 8 = 2048 dims
    float4 a0 = *(const float4*)(b_dec + d0);
    float4 a1 = *(const float4*)(b_dec + d0 + 4);

#pragma unroll 4
    for (int k = 0; k < KTOP; ++k) {
        const float v = sval[k];
        if (v != 0.0f) {
            const float* wr = Wenc + (size_t)sidx[k] * IN_DIM + d0;
            float4 w0 = *(const float4*)(wr);
            float4 w1 = *(const float4*)(wr + 4);
            a0.x = fmaf(v, w0.x, a0.x); a0.y = fmaf(v, w0.y, a0.y);
            a0.z = fmaf(v, w0.z, a0.z); a0.w = fmaf(v, w0.w, a0.w);
            a1.x = fmaf(v, w1.x, a1.x); a1.y = fmaf(v, w1.y, a1.y);
            a1.z = fmaf(v, w1.z, a1.z); a1.w = fmaf(v, w1.w, a1.w);
        }
    }
    *(float4*)(xhat + (size_t)row * IN_DIM + d0)     = a0;
    *(float4*)(xhat + (size_t)row * IN_DIM + d0 + 4) = a1;
}

// ---------------------------------------------------------------------------
extern "C" void kernel_launch(void* const* d_in, const int* in_sizes, int n_in,
                              void* d_out, int out_size, void* d_ws, size_t ws_size,
                              hipStream_t stream) {
    (void)in_sizes; (void)n_in; (void)out_size; (void)ws_size;
    const float* x     = (const float*)d_in[0];
    const float* W_enc = (const float*)d_in[1];
    const float* b_enc = (const float*)d_in[2];
    const float* W_dec = (const float*)d_in[3]; (void)W_dec;  // tied to W_enc^T
    const float* b_dec = (const float*)d_in[4];

    float* xhat = (float*)d_out;
    float* z    = xhat + (size_t)BATCH * IN_DIM;   // outputs concatenated (x_hat, z)

    int*   idx_ws = (int*)d_ws;                                      // 8192*64*4 B
    float* val_ws = (float*)((char*)d_ws + (size_t)BATCH * KTOP * 4); // +8192*64*4 B

    sae_encode<<<dim3(LAT / TN, BATCH / TM), 256, 0, stream>>>(x, W_enc, b_enc, z);

    const size_t topk_lds = (size_t)(LAT + 256) * sizeof(unsigned);
    sae_topk<<<dim3(BATCH), 256, topk_lds, stream>>>(z, idx_ws, val_ws);

    sae_decode<<<dim3(BATCH), 256, 0, stream>>>(W_enc, b_dec, idx_ws, val_ws, xhat);
}
